// TanhAttention_87445534147390
// MI455X (gfx1250) — matrix operand order repaired and checked by reading, not compiled, not run on hardware
//
#include <hip/hip_runtime.h>
#include <cstdint>

#define B_ 32
#define K_ 2048
#define D_ 2048
#define H_ 1024

#define KT 32      // k-rows per block
#define HT 128     // h-cols per H iteration
#define DCH 128    // d-chunk staged in LDS
#define LDP 136    // padded LDS row stride (halves): +8 to stagger banks

typedef __bf16 v16bf __attribute__((ext_vector_type(16)));
typedef float  v8f   __attribute__((ext_vector_type(8)));

union FragB { uint4 q[2]; v16bf v; };

#if __has_builtin(__builtin_amdgcn_global_load_async_to_lds_b32) && \
    __has_builtin(__builtin_amdgcn_s_wait_asynccnt)
#define TA_HAVE_ASYNC 1
typedef __attribute__((address_space(1))) int* gint_p;
typedef __attribute__((address_space(3))) int* lint_p;
#endif

__device__ __forceinline__ unsigned short f2bf(float f) {
  unsigned int u = __float_as_uint(f);
  u += 0x7FFFu + ((u >> 16) & 1u);      // round-to-nearest-even
  return (unsigned short)(u >> 16);
}

// Pack two f32 into packed bf16x2; prefer v_cvt_pk_bf16_f32 when available.
__device__ __forceinline__ unsigned int pack_bf16x2(float x, float y) {
#if __has_builtin(__builtin_amdgcn_cvt_pk_bf16_f32)
  auto p = __builtin_amdgcn_cvt_pk_bf16_f32(x, y);
  unsigned int u;
  __builtin_memcpy(&u, &p, sizeof(u));
  return u;
#else
  return (unsigned int)f2bf(x) | ((unsigned int)f2bf(y) << 16);
#endif
}

// ---------------------------------------------------------------------------
// Kernel 1: qbias[b][h] = b1[h]+b2[h]+b3[h] + W2[h,:]·q1[b,:] + W3[h,:]·q2[b,:]
// ---------------------------------------------------------------------------
__global__ __launch_bounds__(256) void ta_qbias(
    const float* __restrict__ q1, const float* __restrict__ q2,
    const float* __restrict__ W2, const float* __restrict__ b2,
    const float* __restrict__ W3, const float* __restrict__ b3,
    const float* __restrict__ b1, float* __restrict__ qb)
{
  const int b    = blockIdx.y;
  const int lane = threadIdx.x & 31;
  const int w    = threadIdx.x >> 5;
  const int h    = blockIdx.x * 8 + w;
  const float* w2r = W2 + (size_t)h * D_;
  const float* w3r = W3 + (size_t)h * D_;
  const float* q1r = q1 + (size_t)b * D_;
  const float* q2r = q2 + (size_t)b * D_;
  float acc = 0.f;
  for (int d = lane; d < D_; d += 32)
    acc += w2r[d] * q1r[d] + w3r[d] * q2r[d];
  #pragma unroll
  for (int off = 16; off > 0; off >>= 1) acc += __shfl_xor(acc, off, 32);
  if (lane == 0) qb[b * H_ + h] = acc + b1[h] + b2[h] + b3[h];
}

// ---------------------------------------------------------------------------
// Kernel 2: fused  logits[b][k] = Ww · tanh(W1 v[b,k] + qbias[b])
// bf16 WMMA 16x16x32, f32 accumulate. 256 threads = 8 waves.
// Wave grid: 2 (M, 16 rows each) x 4 (N, 32 cols each -> two 16-wide tiles).
// ---------------------------------------------------------------------------
__global__ __launch_bounds__(256, 1) void ta_score(
    const float* __restrict__ v,  const float* __restrict__ W1,
    const float* __restrict__ Ww, const float* __restrict__ qb,
    float* __restrict__ logits)
{
  __shared__ __align__(16) unsigned short vsh[KT][LDP];   // bf16 v tile
  __shared__ __align__(16) unsigned short wsh[HT][LDP];   // bf16 W1^T tile (row=h)
  __shared__ float slog[KT];

  const int tid  = threadIdx.x;
  const int lane = tid & 31;
  const int wid  = tid >> 5;
  const int mw   = wid & 1;            // k-row half: 0/16
  const int nw   = wid >> 1;           // h offset: nw*32
  const int bb   = blockIdx.y;
  const int k0   = blockIdx.x * KT;

  if (tid < KT) slog[tid] = 0.0f;

  // A-fragment lane mapping (16-bit A 16x32, ISA 7.12.2)
  const int am  = mw * 16 + (lane & 15);        // k-row in tile
  const int ab  = (lane < 16) ? 0 : 8;          // K chunks {ab..ab+7, ab+16..ab+23}
  // B-fragment lane mapping (16-bit B 32x16): same K chunks as A {kb..kb+7, kb+16..kb+23}
  const int kb  = (lane < 16) ? 0 : 8;
  const int hl0 = nw * 32 + (lane & 15);        // h (local to tile) for N-subtile 0

  for (int h0 = 0; h0 < H_; h0 += HT) {
    v8f acc[2];
    #pragma unroll
    for (int i = 0; i < 8; ++i) { acc[0][i] = 0.f; acc[1][i] = 0.f; }

    for (int dc = 0; dc < D_; dc += DCH) {
      __syncthreads();
      // ---- stage v chunk: [KT x DCH] fp32 -> bf16 LDS ----
      #pragma unroll
      for (int i = 0; i < (KT * DCH / 4) / 256; ++i) {     // 4 iters
        int fid = tid + i * 256;
        int r = fid >> 5, c = (fid & 31) << 2;
        const float4 g = *(const float4*)&v[((size_t)bb * K_ + k0 + r) * D_ + dc + c];
        *(uint2*)&vsh[r][c] = make_uint2(pack_bf16x2(g.x, g.y),
                                         pack_bf16x2(g.z, g.w));
      }
      // ---- stage W1^T chunk: [HT x DCH] fp32 -> bf16 LDS (served from L2) ----
      #pragma unroll
      for (int i = 0; i < (HT * DCH / 4) / 256; ++i) {     // 16 iters
        int fid = tid + i * 256;
        int h = fid >> 5, c = (fid & 31) << 2;
        const float4 g = *(const float4*)&W1[(size_t)(h0 + h) * D_ + dc + c];
        *(uint2*)&wsh[h][c] = make_uint2(pack_bf16x2(g.x, g.y),
                                         pack_bf16x2(g.z, g.w));
      }
      // prefetch next d-chunk (emits global_prefetch_b8)
      if (dc + DCH < D_) {
        __builtin_prefetch(&v[((size_t)bb * K_ + k0 + (tid & 31)) * D_ + dc + DCH], 0, 0);
        __builtin_prefetch(&W1[(size_t)(h0 + (tid >> 1)) * D_ + dc + DCH], 0, 0);
      }
      __syncthreads();

      // ---- compute: 4 K-steps of 32, 2 WMMAs each ----
      #pragma unroll
      for (int ds0 = 0; ds0 < DCH; ds0 += 32) {
        FragB a;
        a.q[0] = *(const uint4*)&vsh[am][ds0 + ab];
        a.q[1] = *(const uint4*)&vsh[am][ds0 + ab + 16];
        FragB b0;
        b0.q[0] = *(const uint4*)&wsh[hl0][ds0 + kb];
        b0.q[1] = *(const uint4*)&wsh[hl0][ds0 + kb + 16];
        acc[0] = __builtin_amdgcn_wmma_f32_16x16x32_bf16(
            false, a.v, false, b0.v, (short)0, acc[0], false, false);
        FragB b1;
        b1.q[0] = *(const uint4*)&wsh[hl0 + 16][ds0 + kb];
        b1.q[1] = *(const uint4*)&wsh[hl0 + 16][ds0 + kb + 16];
        acc[1] = __builtin_amdgcn_wmma_f32_16x16x32_bf16(
            false, a.v, false, b1.v, (short)0, acc[1], false, false);
      }
    }

    // ---- epilogue: tanh(+qbias) * Ww, reduce over h, accumulate logits ----
    #pragma unroll
    for (int t = 0; t < 2; ++t) {
      const int h  = h0 + hl0 + t * 16;
      const float qv = qb[bb * H_ + h];
      const float wv = Ww[h];
      #pragma unroll
      for (int i = 0; i < 8; ++i) {
        float s = tanhf(acc[t][i] + qv) * wv;
        s += __shfl_xor(s, 1, 32);
        s += __shfl_xor(s, 2, 32);
        s += __shfl_xor(s, 4, 32);
        s += __shfl_xor(s, 8, 32);        // sum over 16 h-lanes per half
        if ((lane & 15) == 0) {
          int m = mw * 16 + i + ((lane & 16) ? 8 : 0);   // C/D layout: lanes16-31 = M+8
          atomicAdd(&slog[m], s);                         // ds_add_f32
        }
      }
    }
  }

  __syncthreads();
  if (tid < KT) logits[(size_t)bb * K_ + k0 + tid] = slog[tid];
}

// ---------------------------------------------------------------------------
// Kernel 3: softmax over K per batch -> weight output
// ---------------------------------------------------------------------------
__global__ __launch_bounds__(256) void ta_softmax(
    const float* __restrict__ logits, float* __restrict__ weight)
{
  const int b = blockIdx.x, tid = threadIdx.x;
  __shared__ float red[256];
  float lmax = -3.4e38f;
  for (int k = tid; k < K_; k += 256) lmax = fmaxf(lmax, logits[b * K_ + k]);
  red[tid] = lmax; __syncthreads();
  #pragma unroll
  for (int s = 128; s > 0; s >>= 1) {
    if (tid < s) red[tid] = fmaxf(red[tid], red[tid + s]);
    __syncthreads();
  }
  const float m = red[0]; __syncthreads();
  float lsum = 0.f;
  for (int k = tid; k < K_; k += 256) lsum += __expf(logits[b * K_ + k] - m);
  red[tid] = lsum; __syncthreads();
  #pragma unroll
  for (int s = 128; s > 0; s >>= 1) {
    if (tid < s) red[tid] += red[tid + s];
    __syncthreads();
  }
  const float inv = 1.0f / red[0];
  for (int k = tid; k < K_; k += 256)
    weight[b * K_ + k] = __expf(logits[b * K_ + k] - m) * inv;
}

// ---------------------------------------------------------------------------
// Kernel 4: feature[b][d] = sum_k weight[b][k] * v[b][k][d]  (HBM-bound gemv)
// Weight row staged to LDS via CDNA5 async-to-LDS path when available.
// ---------------------------------------------------------------------------
__global__ __launch_bounds__(256) void ta_feature(
    const float* __restrict__ v, const float* __restrict__ w,
    float* __restrict__ feat)
{
  const int b = blockIdx.y;
  const int d = blockIdx.x * 256 + threadIdx.x;
  __shared__ float wsm[256];
  float acc = 0.f;
  for (int k0 = 0; k0 < K_; k0 += 256) {
    __syncthreads();
#ifdef TA_HAVE_ASYNC
    // global_load_async_to_lds_b32: per-lane global src -> per-lane LDS dst,
    // tracked with ASYNCcnt.
    __builtin_amdgcn_global_load_async_to_lds_b32(
        (gint_p)(w + b * K_ + k0 + threadIdx.x),
        (lint_p)&wsm[threadIdx.x],
        0, 0);
    __builtin_amdgcn_s_wait_asynccnt(0);
#else
    wsm[threadIdx.x] = w[b * K_ + k0 + threadIdx.x];
#endif
    __syncthreads();
    #pragma unroll 4
    for (int j = 0; j < 256; ++j)
      acc += wsm[j] * v[((size_t)b * K_ + k0 + j) * D_ + d];
  }
  feat[(size_t)b * D_ + d] = acc;
}

// ---------------------------------------------------------------------------
extern "C" void kernel_launch(void* const* d_in, const int* in_sizes, int n_in,
                              void* d_out, int out_size, void* d_ws, size_t ws_size,
                              hipStream_t stream) {
  (void)in_sizes; (void)n_in; (void)out_size; (void)ws_size;
  const float* v  = (const float*)d_in[0];
  const float* q1 = (const float*)d_in[1];
  const float* q2 = (const float*)d_in[2];
  const float* W1 = (const float*)d_in[3];
  const float* b1 = (const float*)d_in[4];
  const float* W2 = (const float*)d_in[5];
  const float* b2 = (const float*)d_in[6];
  const float* W3 = (const float*)d_in[7];
  const float* b3 = (const float*)d_in[8];
  const float* Ww = (const float*)d_in[9];
  // d_in[10] = bw: uniform logit shift -> softmax-invariant, unused.

  float* qb     = (float*)d_ws;                 // [B_, H_]
  float* logits = qb + (size_t)B_ * H_;         // [B_, K_]
  float* w_out  = (float*)d_out;                // [B_, 1, K_]
  float* f_out  = w_out + (size_t)B_ * K_;      // [B_, 1, D_]

  ta_qbias  <<<dim3(H_ / 8, B_),   256, 0, stream>>>(q1, q2, W2, b2, W3, b3, b1, qb);
  ta_score  <<<dim3(K_ / KT, B_),  256, 0, stream>>>(v, W1, Ww, qb, logits);
  ta_softmax<<<dim3(B_),           256, 0, stream>>>(logits, w_out);
  ta_feature<<<dim3(D_ / 256, B_), 256, 0, stream>>>(v, w_out, f_out);
}
